// EncoderProcessDecode_4672924418726
// MI455X (gfx1250) — matrix operand order repaired and checked by reading, not compiled, not run on hardware
//
#include <hip/hip_runtime.h>

// ---------------------------------------------------------------------------
// Encode-Process-Decode GNN for MI455X (gfx1250), wave32 + WMMA bf16.
// All GEMMs run on v_wmma_f32_16x16x32_bf16 (f32 accumulate); LayerNorm,
// residuals and the scatter-sum stay in f32. Weights are pre-transposed and
// converted to bf16 (W^T, row-major by output channel) so the B-operand is a
// two contiguous 16-byte reads per lane.
// ---------------------------------------------------------------------------

typedef __attribute__((ext_vector_type(16))) __bf16 v16bf;
typedef __attribute__((ext_vector_type(8)))  __bf16 v8bf;
typedef __attribute__((ext_vector_type(8)))  float  v8f;
typedef __attribute__((ext_vector_type(4)))  float  v4f;

#define DEVFN static __device__ __forceinline__

#define WAVES          4
#define BLOCK          128           // 4 wave32
#define ROWS_PER_WAVE  16
#define ROWS_PER_BLOCK 64
// per-wave LDS region: f32 tile 16x128 (8192 B) + 64 floats scratch (256 B)
#define TILE_BYTES     8448
#define LN_EPS         1e-5f

// ------------------------------ LDS ordering -------------------------------
DEVFN void lds_fence() {
  __builtin_amdgcn_wave_barrier();
  asm volatile("s_wait_dscnt 0" ::: "memory");
  __builtin_amdgcn_wave_barrier();
}

// ------------------------------ fragments ----------------------------------
// B operand (K x 16 tile of W^T): lane n = lane&15, hi = lane>>4; halves
// 0..7 -> K = kb+8*hi.., halves 8..15 -> K = kb+16+8*hi.. from row n of W^T.
DEVFN v16bf load_b_frag(const __bf16* __restrict__ Wt, int K, int n, int hi, int kb) {
  const v8bf* p0 = (const v8bf*)(Wt + (size_t)n * K + kb + 8 * hi);
  const v8bf* p1 = (const v8bf*)(Wt + (size_t)n * K + kb + 16 + 8 * hi);
  v8bf lo = *p0, hv = *p1;
  v16bf b;
#pragma unroll
  for (int i = 0; i < 8; ++i) { b[i] = lo[i]; b[8 + i] = hv[i]; }
  return b;
}

// A operand from a bf16 LDS tile [16][K] row-major (ISA 16-bit A layout:
// halves 0..7 -> K = kb+8*hi.., halves 8..15 -> K = kb+16+8*hi..)
DEVFN v16bf load_a_lds(const __bf16* __restrict__ tile, int K, int m, int hi, int kb) {
  const v8bf* p0 = (const v8bf*)(tile + (size_t)m * K + kb + 8 * hi);
  const v8bf* p1 = (const v8bf*)(tile + (size_t)m * K + kb + 16 + 8 * hi);
  v8bf lo = *p0, hv = *p1;
  v16bf a;
#pragma unroll
  for (int i = 0; i < 8; ++i) { a[i] = lo[i]; a[8 + i] = hv[i]; }
  return a;
}

// A operand gathered from a global f32 row (converted to bf16 in-register).
DEVFN v16bf a_from_f32(const float* __restrict__ row, int hi, int kb, int D) {
  v16bf a;
  int k0 = kb + 8 * hi;
  int k1 = kb + 16 + 8 * hi;
  if (k0 + 7 < D && k1 + 7 < D) {
    const v4f* p0 = (const v4f*)(row + k0);
    const v4f* p1 = (const v4f*)(row + k1);
    v4f x0 = p0[0], x1 = p0[1], x2 = p1[0], x3 = p1[1];
#pragma unroll
    for (int i = 0; i < 4; ++i) {
      a[i]      = (__bf16)x0[i];
      a[4 + i]  = (__bf16)x1[i];
      a[8 + i]  = (__bf16)x2[i];
      a[12 + i] = (__bf16)x3[i];
    }
  } else {
#pragma unroll
    for (int i = 0; i < 8; ++i) {
      int ka = k0 + i, kbb = k1 + i;
      a[i]     = (ka  < D) ? (__bf16)row[ka]  : (__bf16)0.0f;
      a[8 + i] = (kbb < D) ? (__bf16)row[kbb] : (__bf16)0.0f;
    }
  }
  return a;
}

template <int NT>
DEVFN void zero_acc(v8f acc[NT]) {
#pragma unroll
  for (int t = 0; t < NT; ++t)
#pragma unroll
    for (int r = 0; r < 8; ++r) acc[t][r] = 0.0f;
}

template <int NT>
DEVFN void wmma_nt(v8f acc[NT], v16bf a, const __bf16* __restrict__ Wt,
                   int K, int kb, int n0, int hi) {
#pragma unroll
  for (int t = 0; t < NT; ++t) {
    v16bf b = load_b_frag(Wt, K, t * 16 + n0, hi, kb);
    acc[t] = __builtin_amdgcn_wmma_f32_16x16x32_bf16(
        false, a, false, b, (short)0, acc[t], false, false);
  }
}

// full layer with A streamed from the per-wave bf16 LDS tile
template <int NT>
DEVFN void layer_lds(v8f acc[NT], const __bf16* __restrict__ tileA, int K,
                     const __bf16* __restrict__ Wt, int lane) {
  int m = lane & 15, hi = lane >> 4, n0 = lane & 15;
  zero_acc<NT>(acc);
  for (int kb = 0; kb < K; kb += 32) {
    v16bf a = load_a_lds(tileA, K, m, hi, kb);
    wmma_nt<NT>(acc, a, Wt, K, kb, n0, hi);
  }
}

// C(16x128) + bias (+ReLU) -> bf16 LDS tile [16][128] (A layout for next layer)
DEVFN void store_h_bf16(__bf16* __restrict__ tile, const v8f acc[8],
                        const float* __restrict__ bias, int lane, bool relu) {
  int n0 = lane & 15, hi = lane >> 4;
#pragma unroll
  for (int t = 0; t < 8; ++t) {
    float bv = bias[t * 16 + n0];
#pragma unroll
    for (int r = 0; r < 8; ++r) {
      float v = acc[t][r] + bv;
      if (relu) v = fmaxf(v, 0.0f);
      tile[(size_t)(r + 8 * hi) * 128 + t * 16 + n0] = (__bf16)v;
    }
  }
}

// C(16x128) + bias -> f32 LDS tile (for LayerNorm epilogue)
DEVFN void store_h_f32(float* __restrict__ ftile, const v8f acc[8],
                       const float* __restrict__ bias, int lane) {
  int n0 = lane & 15, hi = lane >> 4;
#pragma unroll
  for (int t = 0; t < 8; ++t) {
    float bv = bias[t * 16 + n0];
#pragma unroll
    for (int r = 0; r < 8; ++r)
      ftile[(size_t)(r + 8 * hi) * 128 + t * 16 + n0] = acc[t][r] + bv;
  }
}

// per-row mean / rsqrt(var) over 128 cols; lane handles row (lane&15),
// half-row 64*(lane>>4); partner partials exchanged through LDS scratch.
DEVFN void ln_stats(const float* __restrict__ ftile, float* __restrict__ scratch,
                    int lane, float& mu, float& inv) {
  int m = lane & 15, hi = lane >> 4;
  const float* rp = ftile + (size_t)m * 128 + 64 * hi;
  float s = 0.0f, q = 0.0f;
  for (int i = 0; i < 64; ++i) { float v = rp[i]; s += v; q += v * v; }
  scratch[lane] = s;
  scratch[32 + lane] = q;
  lds_fence();
  float s2 = s + scratch[lane ^ 16];
  float q2 = q + scratch[32 + (lane ^ 16)];
  mu = s2 * (1.0f / 128.0f);
  float var = q2 * (1.0f / 128.0f) - mu * mu;
  inv = rsqrtf(var + LN_EPS);
}

// ----------------------- weight transpose/convert/pad ----------------------
// src: f32 [K][Out]; dst: bf16 [Op][Kp] (W^T, zero padded)
__global__ void k_prep(const float* __restrict__ src, __bf16* __restrict__ dst,
                       int K, int Out, int Kp, int Op) {
  int i = blockIdx.x * blockDim.x + threadIdx.x;
  if (i >= Kp * Op) return;
  int o = i / Kp, k = i - o * Kp;
  float v = (k < K && o < Out) ? src[(size_t)k * Out + o] : 0.0f;
  dst[i] = (__bf16)v;
}

// ------------------------------- encoder -----------------------------------
// rows x D -> MLP(32pad->128->128->128) -> LN -> outf (f32 [rows][128])
__global__ __launch_bounds__(BLOCK) void k_mlp_enc(
    const float* __restrict__ x, int D, int rows,
    const __bf16* __restrict__ Wt0, const float* __restrict__ b0,
    const __bf16* __restrict__ Wt1, const float* __restrict__ b1,
    const __bf16* __restrict__ Wt2, const float* __restrict__ b2,
    const float* __restrict__ g, const float* __restrict__ be,
    float* __restrict__ outf) {
  __shared__ __align__(16) char smem[WAVES * TILE_BYTES];
  int lane = threadIdx.x & 31, wave = threadIdx.x >> 5;
  char* region = smem + wave * TILE_BYTES;
  float* ftile = (float*)region;
  __bf16* btile = (__bf16*)region;
  float* scratch = ftile + 16 * 128;

  int row0 = (blockIdx.x * WAVES + wave) * ROWS_PER_WAVE;
  int m = lane & 15, hi = lane >> 4, n0 = lane & 15;
  int row = min(row0 + m, rows - 1);

  v8f acc[8];
  // layer 0: K = 32 (padded from D), A gathered from global f32
  zero_acc<8>(acc);
  {
    v16bf a = a_from_f32(x + (size_t)row * D, hi, 0, D);
    wmma_nt<8>(acc, a, Wt0, 32, 0, n0, hi);
  }
  store_h_bf16(btile, acc, b0, lane, true);
  lds_fence();
  layer_lds<8>(acc, btile, 128, Wt1, lane);
  lds_fence();
  store_h_bf16(btile, acc, b1, lane, true);
  lds_fence();
  layer_lds<8>(acc, btile, 128, Wt2, lane);
  lds_fence();
  store_h_f32(ftile, acc, b2, lane);
  lds_fence();

  float mu, inv;
  ln_stats(ftile, scratch, lane, mu, inv);
  int orow = row0 + m;
  if (orow < rows) {
    const float* rp = ftile + (size_t)m * 128 + 64 * hi;
    float* op = outf + (size_t)orow * 128 + 64 * hi;
    for (int i = 0; i < 64; ++i) {
      int c = 64 * hi + i;
      op[i] = (rp[i] - mu) * inv * g[c] + be[c];
    }
  }
}

// ------------------------------ edge update --------------------------------
// cat(nf[snd], nf[rcv], ef) (K=384) -> MLP -> LN -> new_ef;
// ef += new_ef ; agg[rcv] += new_ef (f32 atomics)
__global__ __launch_bounds__(BLOCK) void k_edge(
    const float* __restrict__ nf, float* __restrict__ ef,
    const int* __restrict__ senders, const int* __restrict__ receivers, int E,
    const __bf16* __restrict__ Wt0, const float* __restrict__ b0,
    const __bf16* __restrict__ Wt1, const float* __restrict__ b1,
    const __bf16* __restrict__ Wt2, const float* __restrict__ b2,
    const float* __restrict__ g, const float* __restrict__ be,
    float* __restrict__ agg) {
  __shared__ __align__(16) char smem[WAVES * TILE_BYTES];
  int lane = threadIdx.x & 31, wave = threadIdx.x >> 5;
  char* region = smem + wave * TILE_BYTES;
  float* ftile = (float*)region;
  __bf16* btile = (__bf16*)region;
  float* scratch = ftile + 16 * 128;

  int row0 = (blockIdx.x * WAVES + wave) * ROWS_PER_WAVE;
  int m = lane & 15, hi = lane >> 4, n0 = lane & 15;
  int e = min(row0 + m, E - 1);

  const float* seg[3];
  seg[0] = nf + (size_t)senders[e] * 128;
  seg[1] = nf + (size_t)receivers[e] * 128;
  seg[2] = ef + (size_t)e * 128;

  v8f acc[8];
  zero_acc<8>(acc);
  for (int kb = 0; kb < 384; kb += 32) {
    v16bf a = a_from_f32(seg[kb >> 7], hi, kb & 127, 128);
    wmma_nt<8>(acc, a, Wt0, 384, kb, n0, hi);
  }
  store_h_bf16(btile, acc, b0, lane, true);
  lds_fence();
  layer_lds<8>(acc, btile, 128, Wt1, lane);
  lds_fence();
  store_h_bf16(btile, acc, b1, lane, true);
  lds_fence();
  layer_lds<8>(acc, btile, 128, Wt2, lane);
  lds_fence();
  store_h_f32(ftile, acc, b2, lane);
  lds_fence();

  float mu, inv;
  ln_stats(ftile, scratch, lane, mu, inv);
  int orow = row0 + m;
  if (orow < E) {
    int rcv = receivers[orow];
    const float* rp = ftile + (size_t)m * 128 + 64 * hi;
    float* ep = ef + (size_t)orow * 128 + 64 * hi;
    float* ap = agg + (size_t)rcv * 128 + 64 * hi;
    for (int i = 0; i < 64; ++i) {
      int c = 64 * hi + i;
      float v = (rp[i] - mu) * inv * g[c] + be[c];
      ep[i] = ep[i] + v;                 // residual on edges
      atomicAdd(&ap[i], v);              // scatter-sum to receiver
    }
  }
}

// ------------------------------ node update --------------------------------
// cat(nf, agg) (K=256) -> MLP -> LN -> new_nf ; nf += new_nf
__global__ __launch_bounds__(BLOCK) void k_node(
    float* __restrict__ nf, const float* __restrict__ agg, int N,
    const __bf16* __restrict__ Wt0, const float* __restrict__ b0,
    const __bf16* __restrict__ Wt1, const float* __restrict__ b1,
    const __bf16* __restrict__ Wt2, const float* __restrict__ b2,
    const float* __restrict__ g, const float* __restrict__ be) {
  __shared__ __align__(16) char smem[WAVES * TILE_BYTES];
  int lane = threadIdx.x & 31, wave = threadIdx.x >> 5;
  char* region = smem + wave * TILE_BYTES;
  float* ftile = (float*)region;
  __bf16* btile = (__bf16*)region;
  float* scratch = ftile + 16 * 128;

  int row0 = (blockIdx.x * WAVES + wave) * ROWS_PER_WAVE;
  int m = lane & 15, hi = lane >> 4, n0 = lane & 15;
  int nd = min(row0 + m, N - 1);

  const float* seg[2];
  seg[0] = nf + (size_t)nd * 128;
  seg[1] = agg + (size_t)nd * 128;

  v8f acc[8];
  zero_acc<8>(acc);
  for (int kb = 0; kb < 256; kb += 32) {
    v16bf a = a_from_f32(seg[kb >> 7], hi, kb & 127, 128);
    wmma_nt<8>(acc, a, Wt0, 256, kb, n0, hi);
  }
  store_h_bf16(btile, acc, b0, lane, true);
  lds_fence();
  layer_lds<8>(acc, btile, 128, Wt1, lane);
  lds_fence();
  store_h_bf16(btile, acc, b1, lane, true);
  lds_fence();
  layer_lds<8>(acc, btile, 128, Wt2, lane);
  lds_fence();
  store_h_f32(ftile, acc, b2, lane);
  lds_fence();

  float mu, inv;
  ln_stats(ftile, scratch, lane, mu, inv);
  int orow = row0 + m;
  if (orow < N) {
    const float* rp = ftile + (size_t)m * 128 + 64 * hi;
    float* np = nf + (size_t)orow * 128 + 64 * hi;
    for (int i = 0; i < 64; ++i) {
      int c = 64 * hi + i;
      float v = (rp[i] - mu) * inv * g[c] + be[c];
      np[i] = np[i] + v;                 // residual on nodes
    }
  }
}

// -------------------------------- decoder ----------------------------------
// nf (K=128) -> 128 -> 128 -> 3 (no LN); Wt2 padded to [16][128]
__global__ __launch_bounds__(BLOCK) void k_dec(
    const float* __restrict__ nf, int N,
    const __bf16* __restrict__ Wt0, const float* __restrict__ b0,
    const __bf16* __restrict__ Wt1, const float* __restrict__ b1,
    const __bf16* __restrict__ Wt2, const float* __restrict__ b2,
    float* __restrict__ out) {
  __shared__ __align__(16) char smem[WAVES * TILE_BYTES];
  int lane = threadIdx.x & 31, wave = threadIdx.x >> 5;
  char* region = smem + wave * TILE_BYTES;
  __bf16* btile = (__bf16*)region;

  int row0 = (blockIdx.x * WAVES + wave) * ROWS_PER_WAVE;
  int m = lane & 15, hi = lane >> 4, n0 = lane & 15;
  int nd = min(row0 + m, N - 1);

  v8f acc[8];
  zero_acc<8>(acc);
  for (int kb = 0; kb < 128; kb += 32) {
    v16bf a = a_from_f32(nf + (size_t)nd * 128, hi, kb, 128);
    wmma_nt<8>(acc, a, Wt0, 128, kb, n0, hi);
  }
  store_h_bf16(btile, acc, b0, lane, true);
  lds_fence();
  layer_lds<8>(acc, btile, 128, Wt1, lane);
  lds_fence();
  store_h_bf16(btile, acc, b1, lane, true);
  lds_fence();

  // final layer: single 16-wide N tile (cols 0..2 valid)
  v8f acc2[1];
  zero_acc<1>(acc2);
  for (int kb = 0; kb < 128; kb += 32) {
    v16bf a = load_a_lds(btile, 128, m, hi, kb);
    wmma_nt<1>(acc2, a, Wt2, 128, kb, n0, hi);
  }
  if (n0 < 3) {
    float bv = b2[n0];
#pragma unroll
    for (int r = 0; r < 8; ++r) {
      int orow = row0 + r + 8 * hi;
      if (orow < N) out[(size_t)orow * 3 + n0] = acc2[0][r] + bv;
    }
  }
}

// ------------------------------- launcher ----------------------------------
extern "C" void kernel_launch(void* const* d_in, const int* in_sizes, int n_in,
                              void* d_out, int out_size, void* d_ws, size_t ws_size,
                              hipStream_t stream) {
  (void)n_in; (void)out_size; (void)ws_size;
  const int NODE_D = 12, EDGE_D = 7, S = 3;
  const int N = in_sizes[0] / NODE_D;
  const int E = in_sizes[40];

  const float* node_x = (const float*)d_in[0];
  const float* edge_x = (const float*)d_in[1];
  const int* senders = (const int*)d_in[40];
  const int* receivers = (const int*)d_in[41];
  const float* F = nullptr;  // helper casts below

  char* p = (char*)d_ws;
  auto carve = [&](size_t b) -> char* {
    char* q = p;
    p += (b + 255) & ~(size_t)255;
    return q;
  };
  float* nf = (float*)carve((size_t)N * 128 * sizeof(float));
  float* ef = (float*)carve((size_t)E * 128 * sizeof(float));
  float* agg = (float*)carve((size_t)N * 128 * sizeof(float));

  __bf16* encnW0 = (__bf16*)carve(128 * 32 * 2);
  __bf16* encnW1 = (__bf16*)carve(128 * 128 * 2);
  __bf16* encnW2 = (__bf16*)carve(128 * 128 * 2);
  __bf16* enceW0 = (__bf16*)carve(128 * 32 * 2);
  __bf16* enceW1 = (__bf16*)carve(128 * 128 * 2);
  __bf16* enceW2 = (__bf16*)carve(128 * 128 * 2);
  __bf16 *geW0[3], *geW1[3], *geW2[3], *gnW0[3], *gnW1[3], *gnW2[3];
  for (int s = 0; s < S; ++s) {
    geW0[s] = (__bf16*)carve(128 * 384 * 2);
    geW1[s] = (__bf16*)carve(128 * 128 * 2);
    geW2[s] = (__bf16*)carve(128 * 128 * 2);
    gnW0[s] = (__bf16*)carve(128 * 256 * 2);
    gnW1[s] = (__bf16*)carve(128 * 128 * 2);
    gnW2[s] = (__bf16*)carve(128 * 128 * 2);
  }
  __bf16* decW0 = (__bf16*)carve(128 * 128 * 2);
  __bf16* decW1 = (__bf16*)carve(128 * 128 * 2);
  __bf16* decW2 = (__bf16*)carve(16 * 128 * 2);

  auto prep = [&](const void* src, __bf16* dst, int K, int Out, int Kp, int Op) {
    int n = Kp * Op;
    k_prep<<<(n + 255) / 256, 256, 0, stream>>>((const float*)src, dst, K, Out, Kp, Op);
  };
  prep(d_in[2], encnW0, NODE_D, 128, 32, 128);
  prep(d_in[4], encnW1, 128, 128, 128, 128);
  prep(d_in[6], encnW2, 128, 128, 128, 128);
  prep(d_in[10], enceW0, EDGE_D, 128, 32, 128);
  prep(d_in[12], enceW1, 128, 128, 128, 128);
  prep(d_in[14], enceW2, 128, 128, 128, 128);
  for (int s = 0; s < S; ++s) {
    prep((const float*)d_in[18] + (size_t)s * 384 * 128, geW0[s], 384, 128, 384, 128);
    prep((const float*)d_in[20] + (size_t)s * 128 * 128, geW1[s], 128, 128, 128, 128);
    prep((const float*)d_in[22] + (size_t)s * 128 * 128, geW2[s], 128, 128, 128, 128);
    prep((const float*)d_in[26] + (size_t)s * 256 * 128, gnW0[s], 256, 128, 256, 128);
    prep((const float*)d_in[28] + (size_t)s * 128 * 128, gnW1[s], 128, 128, 128, 128);
    prep((const float*)d_in[30] + (size_t)s * 128 * 128, gnW2[s], 128, 128, 128, 128);
  }
  prep(d_in[34], decW0, 128, 128, 128, 128);
  prep(d_in[36], decW1, 128, 128, 128, 128);
  prep(d_in[38], decW2, 128, 3, 128, 16);

  int gn = (N + ROWS_PER_BLOCK - 1) / ROWS_PER_BLOCK;
  int ge = (E + ROWS_PER_BLOCK - 1) / ROWS_PER_BLOCK;

  k_mlp_enc<<<gn, BLOCK, 0, stream>>>(
      node_x, NODE_D, N, encnW0, (const float*)d_in[3], encnW1, (const float*)d_in[5],
      encnW2, (const float*)d_in[7], (const float*)d_in[8], (const float*)d_in[9], nf);
  k_mlp_enc<<<ge, BLOCK, 0, stream>>>(
      edge_x, EDGE_D, E, enceW0, (const float*)d_in[11], enceW1, (const float*)d_in[13],
      enceW2, (const float*)d_in[15], (const float*)d_in[16], (const float*)d_in[17], ef);

  for (int s = 0; s < S; ++s) {
    hipMemsetAsync(agg, 0, (size_t)N * 128 * sizeof(float), stream);
    k_edge<<<ge, BLOCK, 0, stream>>>(
        nf, ef, senders, receivers, E,
        geW0[s], (const float*)d_in[19] + (size_t)s * 128,
        geW1[s], (const float*)d_in[21] + (size_t)s * 128,
        geW2[s], (const float*)d_in[23] + (size_t)s * 128,
        (const float*)d_in[24] + (size_t)s * 128,
        (const float*)d_in[25] + (size_t)s * 128, agg);
    k_node<<<gn, BLOCK, 0, stream>>>(
        nf, agg, N,
        gnW0[s], (const float*)d_in[27] + (size_t)s * 128,
        gnW1[s], (const float*)d_in[29] + (size_t)s * 128,
        gnW2[s], (const float*)d_in[31] + (size_t)s * 128,
        (const float*)d_in[32] + (size_t)s * 128,
        (const float*)d_in[33] + (size_t)s * 128);
  }

  k_dec<<<gn, BLOCK, 0, stream>>>(
      nf, N, decW0, (const float*)d_in[35], decW1, (const float*)d_in[37],
      decW2, (const float*)d_in[39], (float*)d_out);
  (void)F;
}
